// MultiSageModule_86672440033910
// MI455X (gfx1250) — matrix operand order repaired and checked by reading, not compiled, not run on hardware
//
#include <hip/hip_runtime.h>

// ---------------- problem constants ----------------
#define N_NODES 50000
#define NP      50048          // padded to multiple of 128
#define E_EDGES 800000
#define D       128            // feature dim
#define K3      384            // 3*D (mean|max|min)
#define KC      512            // concat K = 3*D + D
#define G       64             // graphs
#define NL      3              // layers

typedef __attribute__((ext_vector_type(16))) _Float16 v16h;
typedef __attribute__((ext_vector_type(8)))  _Float16 v8h;
typedef __attribute__((ext_vector_type(8)))  float    v8f;

// monotone float<->uint order flip (works for all finite floats, -0<+0)
__device__ __forceinline__ unsigned flipf(float f) {
    unsigned u = __float_as_uint(f);
    return (u & 0x80000000u) ? ~u : (u | 0x80000000u);
}
__device__ __forceinline__ float unflipf(unsigned k) {
    unsigned u = (k & 0x80000000u) ? (k ^ 0x80000000u) : ~k;
    return __uint_as_float(u);
}

// ---------------- degree ----------------
__global__ void deg_zero_kernel(float* __restrict__ deg) {
    int t = blockIdx.x * blockDim.x + threadIdx.x;
    if (t < N_NODES) deg[t] = 0.0f;
}
__global__ void deg_kernel(const int* __restrict__ ei, float* __restrict__ deg) {
    int e = blockIdx.x * blockDim.x + threadIdx.x;
    if (e < E_EDGES) atomicAdd(&deg[ei[E_EDGES + e]], 1.0f);
}

// ---------------- weight prep: Wt[l][n][k] = concat(Wl,Wr) transposed, f16 ----------------
__global__ void prep_w_kernel(const float* __restrict__ Wl, const float* __restrict__ Wr,
                              _Float16* __restrict__ Wt) {
    int t = blockIdx.x * blockDim.x + threadIdx.x;   // NL*D*KC = 196608
    if (t >= NL * D * KC) return;
    int l = t / (D * KC);
    int r = t % (D * KC);
    int n = r / KC;
    int k = r % KC;
    float v;
    if (k < K3) v = Wl[(size_t)l * K3 * D + (size_t)k * D + n];
    else        v = Wr[(size_t)l * D  * D + (size_t)(k - K3) * D + n];
    Wt[(size_t)l * D * KC + (size_t)n * KC + k] = (_Float16)v;
}

// ---------------- per-layer aggregation ----------------
__global__ void init_agg_kernel(float* __restrict__ sum, unsigned* __restrict__ maxk,
                                unsigned* __restrict__ mink) {
    int t = blockIdx.x * blockDim.x + threadIdx.x;   // N*D
    if (t >= N_NODES * D) return;
    sum[t]  = 0.0f;
    maxk[t] = 0u;            // flip(-inf-ish)
    mink[t] = 0xFFFFFFFFu;   // flip(+inf-ish)
}

__global__ void scatter_kernel(const float* __restrict__ h, const int* __restrict__ ei,
                               float* __restrict__ sum, unsigned* __restrict__ maxk,
                               unsigned* __restrict__ mink) {
    int t = blockIdx.x * blockDim.x + threadIdx.x;   // E*128 = 102.4M
    int e = t >> 7;
    int f = t & 127;
    int src = ei[e];
    int dst = ei[E_EDGES + e];
    float v = h[(size_t)src * D + f];
    size_t o = (size_t)dst * D + f;
    atomicAdd(&sum[o], v);
    unsigned k = flipf(v);
    atomicMax(&maxk[o], k);
    atomicMin(&mink[o], k);
}

// build f16 A = [mean | max | min | h]  (rows < N only; pad rows untouched)
__global__ void finalize_kernel(const float* __restrict__ sum, const unsigned* __restrict__ maxk,
                                const unsigned* __restrict__ mink, const float* __restrict__ deg,
                                const float* __restrict__ h, _Float16* __restrict__ A) {
    int t = blockIdx.x * blockDim.x + threadIdx.x;   // N*D
    if (t >= N_NODES * D) return;
    int n = t >> 7;
    int f = t & 127;
    float dg  = deg[n];
    float inv = 1.0f / fmaxf(dg, 1.0f);
    bool  has = dg > 0.0f;
    float mean = sum[t] * inv;
    float mx = has ? unflipf(maxk[t]) : 0.0f;
    float mn = has ? unflipf(mink[t]) : 0.0f;
    size_t base = (size_t)n * KC;
    A[base + f]       = (_Float16)mean;
    A[base + D + f]   = (_Float16)mx;
    A[base + 2*D + f] = (_Float16)mn;
    A[base + K3 + f]  = (_Float16)h[t];
}

// ---------------- WMMA GEMM: hout[N,128] = relu(A[N,512] @ W[512,128] + b) ----------------
// block = 256 threads = 8 wave32; each wave owns a 16-row strip, block covers 128 rows.
// Whole 512x128 f16 weight (128 KB) staged in LDS (CDNA5: 320 KB/WGP).
__global__ void __launch_bounds__(256) gemm_relu_kernel(const _Float16* __restrict__ A,
                                                        const _Float16* __restrict__ Wt,   // [128][512]
                                                        const float* __restrict__ bias,    // [128]
                                                        float* __restrict__ hout) {
    extern __shared__ _Float16 lWt[];   // 128*512 halves = 131072 bytes
    {
        const uint4* s = (const uint4*)Wt;
        uint4*       d = (uint4*)lWt;
        const int n16 = (D * KC * 2) / 16;   // 8192
        for (int i = threadIdx.x; i < n16; i += blockDim.x) d[i] = s[i];
    }
    __syncthreads();

    const int lane  = threadIdx.x & 31;
    const int wave  = threadIdx.x >> 5;
    const int mbase = blockIdx.x * 128 + wave * 16;
    const int arow  = mbase + (lane & 15);
    const int hcol  = (lane >> 4) * 8;    // A-fragment K sub-block select

    v8f acc[8] = {};

    for (int kt = 0; kt < KC / 32; ++kt) {
        // A fragment: 16x32 f16, ISA layout (lane<16: K 0..7 & 16..23; lane>=16: K 8..15 & 24..31)
        v16h a;
        const _Float16* ap = A + (size_t)arow * KC + kt * 32 + hcol;
        *((v8h*)&a)     = *(const v8h*)(ap);
        *((v8h*)&a + 1) = *(const v8h*)(ap + 16);
#pragma unroll
        for (int nt = 0; nt < 8; ++nt) {
            // B fragment from pre-transposed LDS copy: same K layout as A (lane<16: K 0..7 & 16..23; lane>=16: K 8..15 & 24..31)
            const int ncol = nt * 16 + (lane & 15);
            const _Float16* bp = lWt + (size_t)ncol * KC + kt * 32 + (lane >> 4) * 8;
            v16h b;
            *((v8h*)&b)     = *(const v8h*)(bp);
            *((v8h*)&b + 1) = *(const v8h*)(bp + 16);
            acc[nt] = __builtin_amdgcn_wmma_f32_16x16x32_f16(
                false, a, false, b, (short)0, acc[nt], false, false);
        }
    }

    // C/D layout: VGPR r -> row mbase+r+(lane>=16?8:0), col = nt*16 + (lane&15)
#pragma unroll
    for (int nt = 0; nt < 8; ++nt) {
        const int col = nt * 16 + (lane & 15);
        const float bv = bias[col];
#pragma unroll
        for (int r = 0; r < 8; ++r) {
            const int row = mbase + r + (lane >> 4) * 8;
            if (row < N_NODES) {
                float v = acc[nt][r] + bv;
                hout[(size_t)row * D + col] = v > 0.0f ? v : 0.0f;
            }
        }
    }
}

// ---------------- pooling + MLP head ----------------
__global__ void pool_init_kernel(unsigned* __restrict__ pk) {
    int t = blockIdx.x * blockDim.x + threadIdx.x;
    if (t < G * D) pk[t] = 0u;
}
__global__ void pool_scatter_kernel(const float* __restrict__ h, const int* __restrict__ batch,
                                    unsigned* __restrict__ pk) {
    int t = blockIdx.x * blockDim.x + threadIdx.x;   // N*D
    if (t >= N_NODES * D) return;
    int n = t >> 7;
    int f = t & 127;
    atomicMax(&pk[batch[n] * D + f], flipf(h[t]));
}
__global__ void __launch_bounds__(256) mlp_kernel(const unsigned* __restrict__ pk,
                                                  const float* __restrict__ lin_w,  // [128,256]
                                                  const float* __restrict__ lin_b,  // [256]
                                                  const float* __restrict__ out_w,  // [256,10]
                                                  const float* __restrict__ out_b,  // [10]
                                                  float* __restrict__ out) {        // [64,10]
    __shared__ float pl[D];
    __shared__ float hid[256];
    const int g = blockIdx.x;
    const int t = threadIdx.x;
    if (t < D) pl[t] = unflipf(pk[g * D + t]);
    __syncthreads();
    float s = lin_b[t];
    for (int k = 0; k < D; ++k) s += pl[k] * lin_w[k * 256 + t];
    hid[t] = s;
    __syncthreads();
    if (t < 10) {
        float o = out_b[t];
        for (int k = 0; k < 256; ++k) o += hid[k] * out_w[k * 10 + t];
        out[g * 10 + t] = o;
    }
}

// ---------------- launcher ----------------
extern "C" void kernel_launch(void* const* d_in, const int* in_sizes, int n_in,
                              void* d_out, int out_size, void* d_ws, size_t ws_size,
                              hipStream_t stream) {
    (void)in_sizes; (void)n_in; (void)out_size; (void)ws_size;
    const float* x     = (const float*)d_in[0];
    const float* Wl    = (const float*)d_in[1];
    const float* Wr    = (const float*)d_in[2];
    const float* bl    = (const float*)d_in[3];
    const float* lin_w = (const float*)d_in[4];
    const float* lin_b = (const float*)d_in[5];
    const float* out_w = (const float*)d_in[6];
    const float* out_b = (const float*)d_in[7];
    const int*   ei    = (const int*)d_in[8];
    const int*   batch = (const int*)d_in[9];
    float* out = (float*)d_out;

    // carve workspace
    size_t off = 0;
    auto carve = [&](size_t bytes) -> char* {
        char* p = (char*)d_ws + off;
        off += (bytes + 255) & ~(size_t)255;
        return p;
    };
    float*    deg  = (float*)   carve((size_t)NP * 4);
    float*    sum  = (float*)   carve((size_t)NP * D * 4);
    unsigned* maxk = (unsigned*)carve((size_t)NP * D * 4);
    unsigned* mink = (unsigned*)carve((size_t)NP * D * 4);
    _Float16* A16  = (_Float16*)carve((size_t)NP * KC * 2);
    float*    h1   = (float*)   carve((size_t)NP * D * 4);
    float*    h2   = (float*)   carve((size_t)NP * D * 4);
    _Float16* Wt16 = (_Float16*)carve((size_t)NL * D * KC * 2);
    unsigned* pk   = (unsigned*)carve((size_t)G * D * 4);

    const int ND_BLOCKS   = (N_NODES * D) / 256;         // 25000
    const int SC_BLOCKS   = (E_EDGES * 128) / 256;       // 400000
    const int GEMM_BLOCKS = NP / 128;                    // 391
    const size_t LDS_W    = (size_t)D * KC * 2;          // 131072 B

    deg_zero_kernel<<<(N_NODES + 255) / 256, 256, 0, stream>>>(deg);
    deg_kernel<<<(E_EDGES + 255) / 256, 256, 0, stream>>>(ei, deg);
    prep_w_kernel<<<(NL * D * KC) / 256, 256, 0, stream>>>(Wl, Wr, Wt16);

    const float* hin = x;
    float* houts[NL] = { h1, h2, h1 };
    for (int l = 0; l < NL; ++l) {
        init_agg_kernel<<<ND_BLOCKS, 256, 0, stream>>>(sum, maxk, mink);
        scatter_kernel<<<SC_BLOCKS, 256, 0, stream>>>(hin, ei, sum, maxk, mink);
        finalize_kernel<<<ND_BLOCKS, 256, 0, stream>>>(sum, maxk, mink, deg, hin, A16);
        gemm_relu_kernel<<<GEMM_BLOCKS, 256, LDS_W, stream>>>(
            A16, Wt16 + (size_t)l * D * KC, bl + (size_t)l * D, houts[l]);
        hin = houts[l];
    }

    pool_init_kernel<<<(G * D + 255) / 256, 256, 0, stream>>>(pk);
    pool_scatter_kernel<<<ND_BLOCKS, 256, 0, stream>>>(hin, batch, pk);
    mlp_kernel<<<G, 256, 0, stream>>>(pk, lin_w, lin_b, out_w, out_b, out);
}
